// ObservationFusionModule_39230231282387
// MI455X (gfx1250) — matrix operand order repaired and checked by reading, not compiled, not run on hardware
//
#include <hip/hip_runtime.h>
#include <hip/hip_bf16.h>
#include <cstddef>
#include <cstdint>

// ---------------------------------------------------------------------------
// ObservationFusionModule fused kernel for MI455X (gfx1250, wave32, WMMA).
//
//  B=65536 rows, S=4 obs slots (+1 null), H=256, WMLP=128.
//  - All GEMM-shaped work on v_wmma_f32_16x16x32_f16 (f32 accumulate).
//  - Online (flash-style) softmax over the 5 slots.
//  - obs_embeddings (the dominant HBM stream, 256MB) staged into LDS with
//    GLOBAL_LOAD_ASYNC_TO_LDS_B128 one slot ahead (ASYNCcnt-tracked), so the
//    copy overlaps the K/score and V WMMA stages of the previous slot.
// ---------------------------------------------------------------------------

#define B_TOT   65536
#define S_OBS   4
#define H_DIM   256
#define W_MLP   128
#define ROWS    64            // batch rows per workgroup
#define NTHREAD 128           // 4 waves of 32

// dynamic-LDS byte offsets (extern __shared__ base == LDS offset 0: no statics)
#define FUSED_OFF   0         // [64][256] f32   65536
#define HA_OFF      65536     // [64][256] f16   32768
#define QS_OFF      98304     // [64][256] f16   32768
#define MODK_OFF    131072    // [64][256] f16   32768
#define X1_OFF      163840    // [64][128] f16   16384
#define X2_OFF      180224    // [64][128] f16   16384
#define STATS_OFF   196608    //                  4096
#define OBS_OFF     200704    // [64][256] f32   65536
#define SMEM_TOTAL  266240

typedef __attribute__((ext_vector_type(16))) _Float16 v16h;
typedef __attribute__((ext_vector_type(8)))  float    v8f;

// ---------------------------------------------------------------------------
// Weight pre-pack: source W is [N][K] row-major fp32 (out_f x in_f).
// For y = x @ W^T we need B[k][n] = W[n][k] in WMMA B-fragment order:
// fragment f = kt*Ntiles+nt holds 32 lanes x 16 halfs; lane l covers column
// n = nt*16+(l&15), halfs j cover k = kt*32 + (l>>4)*8 + (j&7) + (j>>3)*16.
// ---------------------------------------------------------------------------
__global__ void pack_b_frag(const float* __restrict__ src,
                            _Float16* __restrict__ dst, int K, int N) {
  int tid = blockIdx.x * blockDim.x + threadIdx.x;
  int total = K * N;
  if (tid >= total) return;
  int f    = tid >> 9;
  int rem  = tid & 511;
  int lane = rem >> 4;
  int j    = rem & 15;
  int Ntiles = N >> 4;
  int kt = f / Ntiles;
  int nt = f - kt * Ntiles;
  int n  = nt * 16 + (lane & 15);
  int k  = kt * 32 + ((lane >> 4) << 3) + (j & 7) + ((j >> 3) << 4);
  dst[tid] = (_Float16)src[(size_t)n * K + k];
}

// Null-observation slot K/V projection (row-invariant): computed once.
__global__ void null_kv_kernel(const float* __restrict__ null_emb,
                               const float* __restrict__ Wk, const float* __restrict__ bk,
                               const float* __restrict__ Wv, const float* __restrict__ bv,
                               float* __restrict__ kNull, float* __restrict__ vNull) {
  int o = threadIdx.x;  // 256 threads
  float sk = bk[o], sv = bv[o];
  for (int hh = 0; hh < H_DIM; ++hh) {
    float ne = null_emb[hh];
    sk += ne * Wk[(size_t)o * H_DIM + hh];
    sv += ne * Wv[(size_t)o * H_DIM + hh];
  }
  kNull[o] = sk;
  vNull[o] = sv;
}

// ---------------------------------------------------------------------------
// Fragment loaders
// ---------------------------------------------------------------------------
__device__ __forceinline__ v16h lds_a_frag(const _Float16* buf, int ldk,
                                           int mbase, int kt, int lane) {
  int m  = mbase + (lane & 15);
  int k0 = kt * 32 + ((lane >> 4) << 3);
  const _Float16* p = buf + m * ldk + k0;
  v16h a;
#pragma unroll
  for (int i = 0; i < 8; ++i) { a[i] = p[i]; a[i + 8] = p[i + 16]; }
  return a;
}

__device__ __forceinline__ v16h glb_b_frag(const _Float16* __restrict__ Wp,
                                           int Ntiles, int kt, int nt, int lane) {
  const _Float16* p = Wp + ((size_t)(kt * Ntiles + nt)) * 512 + lane * 16;
  v16h b;
#pragma unroll
  for (int i = 0; i < 16; ++i) b[i] = p[i];
  return b;
}

#define WMMA_F16(a, b, c) \
  __builtin_amdgcn_wmma_f32_16x16x32_f16(false, (a), false, (b), (short)0, (c), false, false)

// Async copy of one obs slot (64 rows x 256 f32 = 64KB) into LDS staging.
// 4096 x 16B chunks; each of 128 threads issues 32 async b128 transfers.
__device__ __forceinline__ void issue_obs_async(const float* __restrict__ obs_emb,
                                                int r0, int s, int tid) {
#pragma unroll 4
  for (int it = 0; it < 32; ++it) {
    int chunk = tid + it * NTHREAD;
    int row   = chunk >> 6;          // 64 chunks per 1KB row
    int col   = (chunk & 63) << 2;   // float index
    const float* g = obs_emb + ((size_t)(r0 + row) * S_OBS + s) * H_DIM + col;
    unsigned int laddr = OBS_OFF + (unsigned)((row * H_DIM + col) * 4);
    unsigned long long ga = (unsigned long long)(uintptr_t)g;
    asm volatile("global_load_async_to_lds_b128 %0, %1, off"
                 :: "v"(laddr), "v"(ga) : "memory");
  }
}

__device__ __forceinline__ void wait_async_all() {
  asm volatile("s_wait_asynccnt 0x0" ::: "memory");
}

// ---------------------------------------------------------------------------
// Main fused kernel: 64 rows per block, 4 waves x 16-row M-tiles.
// ---------------------------------------------------------------------------
__global__ __launch_bounds__(NTHREAD)
void fusion_main(const float* __restrict__ h,
                 const float* __restrict__ obs_emb,
                 const float* __restrict__ stale,
                 const int*   __restrict__ valid,
                 const _Float16* __restrict__ WqP, const _Float16* __restrict__ WkP,
                 const _Float16* __restrict__ WvP, const _Float16* __restrict__ W2P,
                 const _Float16* __restrict__ W3P, const _Float16* __restrict__ WgP,
                 const float* __restrict__ bq, const float* __restrict__ bk,
                 const float* __restrict__ bv,
                 const float* __restrict__ W1, const float* __restrict__ b1,
                 const float* __restrict__ b2, const float* __restrict__ b3,
                 const float* __restrict__ bg,
                 const float* __restrict__ ln_w, const float* __restrict__ ln_b,
                 const float* __restrict__ kNull, const float* __restrict__ vNull,
                 float* __restrict__ out) {
  extern __shared__ __align__(16) char smem_raw[];
  float*    fusedAcc = (float*)(smem_raw + FUSED_OFF);   // [64][256] f32
  _Float16* hA   = (_Float16*)(smem_raw + HA_OFF);       // [64][256] f16
  _Float16* qS   = (_Float16*)(smem_raw + QS_OFF);       // [64][256] f16
  _Float16* modK = (_Float16*)(smem_raw + MODK_OFF);     // [64][256] f16
  _Float16* x1   = (_Float16*)(smem_raw + X1_OFF);       // [64][128] f16
  _Float16* x2   = (_Float16*)(smem_raw + X2_OFF);       // [64][128] f16
  float* stats   = (float*)(smem_raw + STATS_OFF);
  float* obsS    = (float*)(smem_raw + OBS_OFF);         // [64][256] f32 staging
  float* lsv    = stats;        // [64]
  float* scoreB = stats + 64;   // [64]
  float* rowM   = stats + 128;  // [64]
  float* rowD   = stats + 192;  // [64]
  float* rowE   = stats + 256;  // [64]
  float* rowRS  = stats + 320;  // [64]
  float* muA    = stats + 384;  // [64]
  float* rstdA  = stats + 448;  // [64]
  float* anyvA  = stats + 512;  // [64]
  float* pS     = stats + 576;  // [128]
  float* pS2    = stats + 704;  // [128]

  const int tid   = threadIdx.x;
  const int lane  = tid & 31;
  const int wave  = tid >> 5;
  const int mbase = wave * 16;
  const int r0    = blockIdx.x * ROWS;
  const int hi8   = (lane >> 4) << 3;   // 0 or 8: row offset of this lane's acc

  // ---- kick off async staging of obs slot 0 (overlaps h load + q GEMM) -----
  issue_obs_async(obs_emb, r0, 0, tid);

  // ---- Stage h block into LDS as f16 ---------------------------------------
  for (int i = tid; i < ROWS * H_DIM; i += NTHREAD)
    hA[i] = (_Float16)h[(size_t)r0 * H_DIM + i];
  __syncthreads();

  // ---- q = h @ Wq^T + bq  (K=256, N=256) -----------------------------------
  for (int nt = 0; nt < 16; ++nt) {
    v8f acc = {};
#pragma unroll
    for (int kt = 0; kt < 8; ++kt) {
      v16h a = lds_a_frag(hA, H_DIM, mbase, kt, lane);
      v16h b = glb_b_frag(WqP, 16, kt, nt, lane);
      acc = WMMA_F16(a, b, acc);
    }
    int n = nt * 16 + (lane & 15);
    float bqn = bq[n];
#pragma unroll
    for (int r = 0; r < 8; ++r) {
      int row = mbase + hi8 + r;
      qS[row * H_DIM + n] = (_Float16)(acc[r] + bqn);
    }
  }
  __syncthreads();

  // ---- Null slot initializes the online softmax ----------------------------
  for (int i = tid; i < ROWS * H_DIM; i += NTHREAD)
    fusedAcc[i] = vNull[i & (H_DIM - 1)];
  if (tid < ROWS) {
    int row = tid;
    float sdot = 0.f;
    for (int n = 0; n < H_DIM; ++n)
      sdot += (float)qS[row * H_DIM + n] * kNull[n];
    rowM[row] = sdot * (1.0f / 16.0f);   // /sqrt(H)
    rowD[row] = 1.0f;
    int av = 0;
    for (int s = 0; s < S_OBS; ++s) av |= valid[(size_t)(r0 + row) * S_OBS + s];
    anyvA[row] = av ? 1.0f : 0.0f;
  }
  __syncthreads();

  // ---- Stream the 4 real observation slots ---------------------------------
  for (int s = 0; s < S_OBS; ++s) {
    // staleness scalar
    if (tid < ROWS)
      lsv[tid] = log1pf(stale[(size_t)(r0 + tid) * S_OBS + s]);
    __syncthreads();

    // layer1: x1 = relu(ls*W1 + b1)   [64][128]  (outer product, VALU)
    for (int i = tid; i < ROWS * W_MLP; i += NTHREAD) {
      int row = i >> 7, w = i & 127;
      x1[i] = (_Float16)fmaxf(lsv[row] * W1[w] + b1[w], 0.f);
    }
    __syncthreads();

    // layer2: x2 = relu(x1 @ W2^T + b2)   (K=128, N=128)
    for (int nt = 0; nt < 8; ++nt) {
      v8f acc = {};
#pragma unroll
      for (int kt = 0; kt < 4; ++kt) {
        v16h a = lds_a_frag(x1, W_MLP, mbase, kt, lane);
        v16h b = glb_b_frag(W2P, 8, kt, nt, lane);
        acc = WMMA_F16(a, b, acc);
      }
      int n = nt * 16 + (lane & 15);
      float b2n = b2[n];
#pragma unroll
      for (int r = 0; r < 8; ++r) {
        int row = mbase + hi8 + r;
        x2[row * W_MLP + n] = (_Float16)fmaxf(acc[r] + b2n, 0.f);
      }
    }

    // make sure this slot's async obs staging has fully landed in LDS
    wait_async_all();
    __syncthreads();

    // layer3 + modulate: modK = f16(obsS + x2 @ W3^T + b3)  (K=128, N=256)
    for (int nt = 0; nt < 16; ++nt) {
      v8f acc = {};
#pragma unroll
      for (int kt = 0; kt < 4; ++kt) {
        v16h a = lds_a_frag(x2, W_MLP, mbase, kt, lane);
        v16h b = glb_b_frag(W3P, 16, kt, nt, lane);
        acc = WMMA_F16(a, b, acc);
      }
      int n = nt * 16 + (lane & 15);
      float b3n = b3[n];
#pragma unroll
      for (int r = 0; r < 8; ++r) {
        int row = mbase + hi8 + r;
        float ob = obsS[row * H_DIM + n];
        modK[row * H_DIM + n] = (_Float16)(ob + acc[r] + b3n);
      }
    }
    __syncthreads();

    // staging buffer free again: prefetch next slot (overlaps k/score + v)
    if (s + 1 < S_OBS)
      issue_obs_async(obs_emb, r0, s + 1, tid);

    // k projection fused with q·k score (never materialize k)
    float part[8];
#pragma unroll
    for (int r = 0; r < 8; ++r) part[r] = 0.f;
    for (int nt = 0; nt < 16; ++nt) {
      v8f acc = {};
#pragma unroll
      for (int kt = 0; kt < 8; ++kt) {
        v16h a = lds_a_frag(modK, H_DIM, mbase, kt, lane);
        v16h b = glb_b_frag(WkP, 16, kt, nt, lane);
        acc = WMMA_F16(a, b, acc);
      }
      int n = nt * 16 + (lane & 15);
      float bkn = bk[n];
#pragma unroll
      for (int r = 0; r < 8; ++r) {
        int row = mbase + hi8 + r;
        part[r] += (acc[r] + bkn) * (float)qS[row * H_DIM + n];
      }
    }
#pragma unroll
    for (int r = 0; r < 8; ++r) {   // reduce across the 16-lane N group
      float v = part[r];
      v += __shfl_xor(v, 1, 32);
      v += __shfl_xor(v, 2, 32);
      v += __shfl_xor(v, 4, 32);
      v += __shfl_xor(v, 8, 32);
      part[r] = v;
    }
    int sub = lane & 15;
#pragma unroll
    for (int r = 0; r < 8; ++r)
      if (sub == r) scoreB[mbase + hi8 + r] = part[r];
    __syncthreads();

    // online-softmax state update (one thread per row)
    if (tid < ROWS) {
      int row = tid;
      float sc = scoreB[row] * (1.0f / 16.0f);
      if (!valid[(size_t)(r0 + row) * S_OBS + s]) sc = -1e9f;
      float m0 = rowM[row];
      float mn = fmaxf(m0, sc);
      float e  = __expf(sc - mn);
      float rs = __expf(m0 - mn);
      rowD[row] = rowD[row] * rs + e;
      rowM[row] = mn;
      rowE[row] = e;
      rowRS[row] = rs;
    }
    __syncthreads();

    // v projection + streamed accumulation into fusedAcc
    for (int nt = 0; nt < 16; ++nt) {
      v8f acc = {};
#pragma unroll
      for (int kt = 0; kt < 8; ++kt) {
        v16h a = lds_a_frag(modK, H_DIM, mbase, kt, lane);
        v16h b = glb_b_frag(WvP, 16, kt, nt, lane);
        acc = WMMA_F16(a, b, acc);
      }
      int n = nt * 16 + (lane & 15);
      float bvn = bv[n];
#pragma unroll
      for (int r = 0; r < 8; ++r) {
        int row = mbase + hi8 + r;
        int idx = row * H_DIM + n;
        fusedAcc[idx] = fusedAcc[idx] * rowRS[row] + rowE[row] * (acc[r] + bvn);
      }
    }
    __syncthreads();
  }

  // ---- finalize softmax, stash f16 copy of fused for the gate GEMM ---------
  for (int i = tid; i < ROWS * H_DIM; i += NTHREAD) {
    int row = i >> 8;
    float f = fusedAcc[i] / rowD[row];
    fusedAcc[i] = f;
    modK[i] = (_Float16)f;
  }
  __syncthreads();

  // layernorm statistics (2 threads per row)
  {
    int row = tid >> 1;
    int base = row * H_DIM + (tid & 1) * 128;
    float sm = 0.f, sq = 0.f;
    for (int j = 0; j < 128; ++j) {
      float v = fusedAcc[base + j];
      sm += v; sq += v * v;
    }
    pS[tid] = sm; pS2[tid] = sq;
  }
  __syncthreads();
  if (tid < ROWS) {
    float sm = pS[2 * tid] + pS[2 * tid + 1];
    float sq = pS2[2 * tid] + pS2[2 * tid + 1];
    float mu = sm * (1.0f / H_DIM);
    float var = sq * (1.0f / H_DIM) - mu * mu;
    muA[tid] = mu;
    rstdA[tid] = rsqrtf(var + 1e-5f);
  }
  __syncthreads();

  // ---- gate GEMM (K=512: [h | fused]) + LN + gated residual + store --------
  for (int nt = 0; nt < 16; ++nt) {
    v8f acc = {};
#pragma unroll
    for (int kt = 0; kt < 16; ++kt) {
      v16h a = (kt < 8) ? lds_a_frag(hA, H_DIM, mbase, kt, lane)
                        : lds_a_frag(modK, H_DIM, mbase, kt - 8, lane);
      v16h b = glb_b_frag(WgP, 16, kt, nt, lane);
      acc = WMMA_F16(a, b, acc);
    }
    int n = nt * 16 + (lane & 15);
    float bgn = bg[n], lw = ln_w[n], lb = ln_b[n];
#pragma unroll
    for (int r = 0; r < 8; ++r) {
      int row = mbase + hi8 + r;
      float g   = 1.0f / (1.0f + __expf(-(acc[r] + bgn)));
      float eff = anyvA[row] * g;
      float f   = fusedAcc[row * H_DIM + n];
      float up  = (f - muA[row]) * rstdA[row] * lw + lb;
      float hv  = (float)hA[row * H_DIM + n];
      out[(size_t)(r0 + row) * H_DIM + n] = hv + eff * up;
    }
  }
}

// ---------------------------------------------------------------------------
// Host launcher
// ---------------------------------------------------------------------------
extern "C" void kernel_launch(void* const* d_in, const int* in_sizes, int n_in,
                              void* d_out, int out_size, void* d_ws, size_t ws_size,
                              hipStream_t stream) {
  const float* h        = (const float*)d_in[0];
  const float* obs_emb  = (const float*)d_in[1];
  const float* stale    = (const float*)d_in[2];
  const int*   valid    = (const int*)d_in[3];
  const float* null_emb = (const float*)d_in[4];
  const float* Wq = (const float*)d_in[5];  const float* bq = (const float*)d_in[6];
  const float* Wk = (const float*)d_in[7];  const float* bk = (const float*)d_in[8];
  const float* Wv = (const float*)d_in[9];  const float* bv = (const float*)d_in[10];
  const float* W1 = (const float*)d_in[11]; const float* b1 = (const float*)d_in[12];
  const float* W2 = (const float*)d_in[13]; const float* b2 = (const float*)d_in[14];
  const float* W3 = (const float*)d_in[15]; const float* b3 = (const float*)d_in[16];
  const float* Wg = (const float*)d_in[17]; const float* bg = (const float*)d_in[18];
  const float* ln_w = (const float*)d_in[19];
  const float* ln_b = (const float*)d_in[20];
  float* out = (float*)d_out;

  // workspace carve (halfs, then floats)
  _Float16* wsH = (_Float16*)d_ws;
  _Float16* WqP = wsH;
  _Float16* WkP = wsH + 65536;
  _Float16* WvP = wsH + 131072;
  _Float16* W2P = wsH + 196608;
  _Float16* W3P = wsH + 212992;
  _Float16* WgP = wsH + 245760;
  float* kNull = (float*)((char*)d_ws + 753664);
  float* vNull = kNull + 256;

  // pack weights into WMMA B-fragment order (f16)
  pack_b_frag<<<(256 * 256 + 255) / 256, 256, 0, stream>>>(Wq, WqP, 256, 256);
  pack_b_frag<<<(256 * 256 + 255) / 256, 256, 0, stream>>>(Wk, WkP, 256, 256);
  pack_b_frag<<<(256 * 256 + 255) / 256, 256, 0, stream>>>(Wv, WvP, 256, 256);
  pack_b_frag<<<(128 * 128 + 255) / 256, 256, 0, stream>>>(W2, W2P, 128, 128);
  pack_b_frag<<<(128 * 256 + 255) / 256, 256, 0, stream>>>(W3, W3P, 128, 256);
  pack_b_frag<<<(512 * 256 + 255) / 256, 256, 0, stream>>>(Wg, WgP, 512, 256);

  null_kv_kernel<<<1, 256, 0, stream>>>(null_emb, Wk, bk, Wv, bv, kNull, vNull);

  fusion_main<<<B_TOT / ROWS, NTHREAD, SMEM_TOTAL, stream>>>(
      h, obs_emb, stale, valid,
      WqP, WkP, WvP, W2P, W3P, WgP,
      bq, bk, bv, W1, b1, b2, b3, bg, ln_w, ln_b,
      kNull, vNull, out);
}
